// EKV_ResNet_44822278701750
// MI455X (gfx1250) — matrix operand order repaired and checked by reading, not compiled, not run on hardware
//
#include <hip/hip_runtime.h>
#include <hip/hip_bf16.h>

// ---------------------------------------------------------------------------
// EKV-ResNet forward for MI455X (gfx1250, wave32).
//   - EKV "conv" nonlinearity is per (input,weight) pair -> not bilinear ->
//     TRANS-pipe bound (v_exp_f32/v_log_f32), theta staged in LDS via
//     GLOBAL_LOAD_ASYNC_TO_LDS (ASYNCcnt) when available.
//   - BN (training-mode batch stats) via per-channel LDS tree reduction.
//   - Final FC = the only true GEMM -> v_wmma_f32_16x16x32_f16, with the
//     weight tile pre-staged zero-padded in LDS so fragment build is
//     branch-free (EXEC stays all-ones around WMMA).
// ---------------------------------------------------------------------------

typedef __attribute__((ext_vector_type(16))) _Float16 v16h;
typedef __attribute__((ext_vector_type(8)))  float    v8f;

#define EKV_INV_DENOM 13.3333333333f   /* 1 / (2*1.5*0.025) */
#define EKV_VD_SCALED 1.3333333333f    /* 0.1 / 0.075        */
#define EKV_OUT_SCALE 28.125f          /* R_TIA * EKV_ALPHA  */

#if defined(__has_builtin)
#if __has_builtin(__builtin_amdgcn_global_load_async_to_lds_b32)
#define HAVE_ASYNC_TO_LDS 1
#endif
#endif

#define AS1 __attribute__((address_space(1)))
#define AS3 __attribute__((address_space(3)))

__device__ __forceinline__ float softplus_f(float z) {
  // numerically stable: max(z,0) + log1p(exp(-|z|)); z can reach ~±500 here
  return fmaxf(z, 0.0f) + __logf(1.0f + __expf(-fabsf(z)));
}

// One block per (n, o). Theta slice for o cached in LDS (async DMA path on
// gfx1250); threads sweep output pixels. KT is 3 or 1 (compile-time unroll).
template <int KT>
__global__ void ekv_conv_kernel(const float* __restrict__ in,
                                const float* __restrict__ theta,
                                float* __restrict__ out,
                                int N, int Cin, int H, int W, int O,
                                int stride, int pad,
                                int Ho, int Wo, float inScale) {
  __shared__ float sTheta[576];  // up to 64*3*3 floats
  const int n = blockIdx.x / O;
  const int o = blockIdx.x % O;
  const int tk = Cin * KT * KT;
  const float* gTheta = theta + o * tk;
#ifdef HAVE_ASYNC_TO_LDS
  for (int i = threadIdx.x; i < tk; i += blockDim.x)
    __builtin_amdgcn_global_load_async_to_lds_b32(
        (AS1 int*)(gTheta + i), (AS3 int*)(&sTheta[i]), 0, 0);
#if __has_builtin(__builtin_amdgcn_s_wait_asynccnt)
  __builtin_amdgcn_s_wait_asynccnt(0);
#else
  asm volatile("s_wait_asynccnt 0x0" ::: "memory");
#endif
#else
  for (int i = threadIdx.x; i < tk; i += blockDim.x)
    sTheta[i] = gTheta[i];
#endif
  __syncthreads();

  const float* inN = in + (size_t)n * Cin * H * W;
  __builtin_prefetch(inN + threadIdx.x, 0, 0);  // warm input image (L2)
  const int HWo = Ho * Wo;
  for (int p = threadIdx.x; p < HWo; p += blockDim.x) {
    const int ho = p / Wo, wo = p % Wo;
    const int hi0 = ho * stride - pad;
    const int wi0 = wo * stride - pad;
    float acc = 0.0f;
    for (int c = 0; c < Cin; ++c) {
      const float* inc = inN + c * H * W;
      const float* th  = sTheta + c * KT * KT;
#pragma unroll
      for (int ki = 0; ki < KT; ++ki) {
        const int hi = hi0 + ki;
        const bool hok = ((unsigned)hi < (unsigned)H);
#pragma unroll
        for (int kj = 0; kj < KT; ++kj) {
          const int wi = wi0 + kj;
          float v = 0.0f;  // zero-padded voltage (reference pads V with 0)
          if (hok && ((unsigned)wi < (unsigned)W)) v = inScale * inc[hi * W + wi];
          const float z1 = (v - th[ki * KT + kj]) * EKV_INV_DENOM;
          const float z2 = z1 - EKV_VD_SCALED;
          const float s1 = softplus_f(z1);
          const float s2 = softplus_f(z2);
          acc += (s1 - s2) * (s1 + s2);   // s1^2 - s2^2
        }
      }
    }
    out[((size_t)(n * O + o)) * HWo + p] = EKV_OUT_SCALE * acc;
  }
}

// One block per channel: biased batch stats over (N,H,W).
__global__ void bn_stats_kernel(const float* __restrict__ x, int N, int C, int HW,
                                float* __restrict__ mean, float* __restrict__ invstd) {
  __shared__ float sSum[256];
  __shared__ float sSq[256];
  const int c = blockIdx.x;
  float s = 0.0f, q = 0.0f;
  for (int n = 0; n < N; ++n) {
    const float* xc = x + ((size_t)(n * C + c)) * HW;
    for (int p = threadIdx.x; p < HW; p += blockDim.x) {
      const float v = xc[p];
      s += v;
      q += v * v;
    }
  }
  sSum[threadIdx.x] = s;
  sSq[threadIdx.x]  = q;
  __syncthreads();
  for (int off = blockDim.x >> 1; off > 0; off >>= 1) {
    if ((int)threadIdx.x < off) {
      sSum[threadIdx.x] += sSum[threadIdx.x + off];
      sSq[threadIdx.x]  += sSq[threadIdx.x + off];
    }
    __syncthreads();
  }
  if (threadIdx.x == 0) {
    const float total = (float)(N * HW);
    const float m   = sSum[0] / total;
    const float var = sSq[0] / total - m * m;
    mean[c]   = m;
    invstd[c] = rsqrtf(var + 1e-5f);
  }
}

// Elementwise normalize (+optional residual). Safe in-place (out may == x).
__global__ void bn_apply_kernel(const float* __restrict__ x,
                                const float* __restrict__ mean,
                                const float* __restrict__ invstd,
                                const float* __restrict__ gamma,
                                const float* __restrict__ beta,
                                const float* __restrict__ res,
                                float* __restrict__ out, int N, int C, int HW) {
  const int total = N * C * HW;
  for (int i = blockIdx.x * blockDim.x + threadIdx.x; i < total;
       i += gridDim.x * blockDim.x) {
    const int c = (i / HW) % C;
    float v = (x[i] - mean[c]) * invstd[c] * gamma[c] + beta[c];
    if (res) v += res[i];
    out[i] = v;
  }
}

__global__ void avgpool_kernel(const float* __restrict__ x, float* __restrict__ out,
                               int NC, int HW) {
  const int i = blockIdx.x * blockDim.x + threadIdx.x;
  if (i < NC) {
    float s = 0.0f;
    const float* p = x + (size_t)i * HW;
    for (int j = 0; j < HW; ++j) s += p[j];
    out[i] = s / (float)HW;
  }
}

// FC: out[32,10] = pooled[32,64] @ W[64,10] + b.  Two full wave32 waves,
// each owns one 16-row tile; K=64 as two v_wmma_f32_16x16x32_f16 steps.
// W is staged zero-padded in LDS so the B-fragment build is branch-free
// (EXEC all-ones throughout, as WMMA requires).
// Fragment packing per CDNA5 ISA 7.12.2 wave32 layouts.
__global__ void fc_wmma_kernel(const float* __restrict__ A,     // [32,64]
                               const float* __restrict__ Wt,    // [64,10]
                               const float* __restrict__ bias,  // [10]
                               float* __restrict__ out) {       // [32,10]
  __shared__ _Float16 sB[64][16];  // zero-padded to N=16
  __shared__ float sBias[16];
  for (int i = threadIdx.x; i < 64 * 16; i += blockDim.x) {
    const int k = i >> 4, nn = i & 15;
    sB[k][nn] = (_Float16)((nn < 10) ? Wt[k * 10 + nn] : 0.0f);
  }
  if (threadIdx.x < 16)
    sBias[threadIdx.x] = (threadIdx.x < 10) ? bias[threadIdx.x] : 0.0f;
  __syncthreads();

  const int lane = threadIdx.x & 31;
  const int wave = threadIdx.x >> 5;
  const int m0 = wave * 16;
  const int mr = m0 + (lane & 15);
  const int nn = lane & 15;
  const int ah = (lane >> 4) * 8;    // A K-half offset
  const int bh = (lane >> 4) * 8;    // B K-half offset
  v8f c = {};
  for (int ks = 0; ks < 64; ks += 32) {
    v16h a, b;
    // A 16x32 f16: lane holds row m = lane%16; VGPR0..3 -> K = ah+{0..7},
    // VGPR4..7 -> K = 16+ah+{0..7}
#pragma unroll
    for (int v = 0; v < 8; ++v) {
      const int k = ks + ((v < 4) ? (ah + 2 * v) : (16 + ah + 2 * (v - 4)));
      a[2 * v]     = (_Float16)A[mr * 64 + k];
      a[2 * v + 1] = (_Float16)A[mr * 64 + k + 1];
    }
    // B 32x16 f16: lane holds col n = lane%16; VGPR0..3 -> K = bh+{0..7}, VGPR4..7 -> K = 16+bh+{0..7}
#pragma unroll
    for (int v = 0; v < 8; ++v) {
      const int k = ks + ((v < 4) ? (bh + 2 * v) : (16 + bh + 2 * (v - 4)));
      b[2 * v]     = sB[k][nn];
      b[2 * v + 1] = sB[k + 1][nn];
    }
    c = __builtin_amdgcn_wmma_f32_16x16x32_f16(false, a, false, b,
                                               (short)0, c, false, false);
  }
  // D layout: lane holds col n = lane%16; VGPR r -> row r + 8*(lane/16)
  const int mh = (lane >> 4) * 8;
  if (nn < 10) {
#pragma unroll
    for (int r = 0; r < 8; ++r)
      out[(m0 + mh + r) * 10 + nn] = c[r] + sBias[nn];
  }
}

// ---------------------------------------------------------------------------
extern "C" void kernel_launch(void* const* d_in, const int* in_sizes, int n_in,
                              void* d_out, int out_size, void* d_ws, size_t ws_size,
                              hipStream_t stream) {
  (void)out_size; (void)ws_size;
  // ---- identify inputs by flat size (unique sizes) ----
  int i16[6], c16 = 0, i32a[6], c32 = 0, i64a[6], c64 = 0, i2304[2], c2304 = 0;
  int iX = -1, iTop = -1, iB2c1 = -1, iB2c2 = -1, iB2sc = -1;
  int iB3c1 = -1, iB3c2 = -1, iB3sc = -1, iFw = -1, iFb = -1;
  for (int i = 0; i < n_in; ++i) {
    switch (in_sizes[i]) {
      case 32768: iX = i; break;
      case 432:   iTop = i; break;
      case 2304:  if (c2304 < 2) i2304[c2304++] = i; break;
      case 4608:  iB2c1 = i; break;
      case 9216:  iB2c2 = i; break;
      case 512:   iB2sc = i; break;
      case 18432: iB3c1 = i; break;
      case 36864: iB3c2 = i; break;
      case 2048:  iB3sc = i; break;
      case 640:   iFw = i; break;
      case 10:    iFb = i; break;
      case 16:    if (c16 < 6) i16[c16++] = i; break;
      case 32:    if (c32 < 6) i32a[c32++] = i; break;
      case 64:    if (c64 < 6) i64a[c64++] = i; break;
      default: break;
    }
  }
  if (iX < 0 || iTop < 0 || c2304 < 2 || iB2c1 < 0 || iB2c2 < 0 || iB2sc < 0 ||
      iB3c1 < 0 || iB3c2 < 0 || iB3sc < 0 || iFw < 0 || iFb < 0 ||
      c16 < 6 || c32 < 6 || c64 < 6)
    return;

  // insertion-order flatten starts [x, conv1_theta(432), ...]; jax tree
  // flatten is alphabetical (betas before gammas). Disambiguate:
  const bool ins = (n_in > 1 && in_sizes[0] == 32768 && in_sizes[1] == 432);
  int topG, topB, b1g1, b1b1, b1g2, b1b2;
  int b2g1, b2b1, b2g2, b2b2, b2scg, b2scb;
  int b3g1, b3b1, b3g2, b3b2, b3scg, b3scb;
  if (ins) {  // g before b, top-level bn first
    topG = i16[0]; topB = i16[1]; b1g1 = i16[2]; b1b1 = i16[3]; b1g2 = i16[4]; b1b2 = i16[5];
    b2g1 = i32a[0]; b2b1 = i32a[1]; b2g2 = i32a[2]; b2b2 = i32a[3]; b2scg = i32a[4]; b2scb = i32a[5];
    b3g1 = i64a[0]; b3b1 = i64a[1]; b3g2 = i64a[2]; b3b2 = i64a[3]; b3scg = i64a[4]; b3scb = i64a[5];
  } else {    // alphabetical: b before g, block bns before top bn
    b1b1 = i16[0]; b1g1 = i16[1]; b1b2 = i16[2]; b1g2 = i16[3]; topB = i16[4]; topG = i16[5];
    b2b1 = i32a[0]; b2g1 = i32a[1]; b2b2 = i32a[2]; b2g2 = i32a[3]; b2scb = i32a[4]; b2scg = i32a[5];
    b3b1 = i64a[0]; b3g1 = i64a[1]; b3b2 = i64a[2]; b3g2 = i64a[3]; b3scb = i64a[4]; b3scg = i64a[5];
  }
  const int iB1c1 = i2304[0], iB1c2 = i2304[1];

  const float* X = (const float*)d_in[iX];
  auto P = [&](int i) { return (const float*)d_in[i]; };

  // ---- workspace layout (floats) ----
  float* bufX = (float*)d_ws;          // block input / h1 scratch (524288)
  float* bufA = bufX + 524288;         // conv output
  float* bufB = bufA + 524288;         // bn output / block output
  float* bufC = bufB + 524288;         // shortcut
  float* stat = bufC + 524288;         // mean[64] + invstd[64]
  float* pooled = stat + 128;          // [32,64]
  float* meanP = stat, *invP = stat + 64;

  auto conv = [&](const float* in, const float* th, float* o_, int N, int Cin,
                  int H, int W, int O, int K, int s, int p, float sc) {
    const int Ho = (H + 2 * p - K) / s + 1, Wo = (W + 2 * p - K) / s + 1;
    const int thr = (Ho * Wo >= 256) ? 256 : 64;
    if (K == 3)
      ekv_conv_kernel<3><<<dim3(N * O), dim3(thr), 0, stream>>>(
          in, th, o_, N, Cin, H, W, O, s, p, Ho, Wo, sc);
    else
      ekv_conv_kernel<1><<<dim3(N * O), dim3(thr), 0, stream>>>(
          in, th, o_, N, Cin, H, W, O, s, p, Ho, Wo, sc);
  };
  auto bnstats = [&](const float* xp, int N, int C, int HW) {
    bn_stats_kernel<<<dim3(C), dim3(256), 0, stream>>>(xp, N, C, HW, meanP, invP);
  };
  auto bnapply = [&](const float* xp, int gi, int bi, const float* res, float* o_,
                     int N, int C, int HW) {
    const int total = N * C * HW;
    int blocks = (total + 255) / 256;
    if (blocks > 2048) blocks = 2048;
    bn_apply_kernel<<<dim3(blocks), dim3(256), 0, stream>>>(
        xp, meanP, invP, P(gi), P(bi), res, o_, N, C, HW);
  };

  const int N = 32;

  // ---- stem: h = BN(R*ekv_conv(5x, theta1, s=1, p=1)) ----
  conv(X, P(iTop), bufA, N, 3, 32, 32, 16, 3, 1, 1, 5.0f);
  bnstats(bufA, N, 16, 1024);
  bnapply(bufA, topG, topB, nullptr, bufX, N, 16, 1024);        // bufX = h

  // ---- block1: 16->16, 32x32, identity shortcut ----
  conv(bufX, P(iB1c1), bufA, N, 16, 32, 32, 16, 3, 1, 1, 1.0f);
  bnstats(bufA, N, 16, 1024);
  bnapply(bufA, b1g1, b1b1, nullptr, bufB, N, 16, 1024);        // h1
  conv(bufB, P(iB1c2), bufA, N, 16, 32, 32, 16, 3, 1, 1, 1.0f);
  bnstats(bufA, N, 16, 1024);
  bnapply(bufA, b1g2, b1b2, bufX, bufB, N, 16, 1024);           // bufB = out1

  // ---- block2: 16->32, stride 2 -> 16x16, conv shortcut ----
  conv(bufB, P(iB2c1), bufA, N, 16, 32, 32, 32, 3, 2, 1, 1.0f);
  bnstats(bufA, N, 32, 256);
  bnapply(bufA, b2g1, b2b1, nullptr, bufX, N, 32, 256);         // h1
  conv(bufX, P(iB2c2), bufA, N, 32, 16, 16, 32, 3, 1, 1, 1.0f);
  conv(bufB, P(iB2sc), bufC, N, 16, 32, 32, 32, 1, 2, 0, 1.0f); // shortcut
  bnstats(bufC, N, 32, 256);
  bnapply(bufC, b2scg, b2scb, nullptr, bufC, N, 32, 256);       // in-place
  bnstats(bufA, N, 32, 256);
  bnapply(bufA, b2g2, b2b2, bufC, bufB, N, 32, 256);            // bufB = out2

  // ---- block3: 32->64, stride 2 -> 8x8, conv shortcut ----
  conv(bufB, P(iB3c1), bufA, N, 32, 16, 16, 64, 3, 2, 1, 1.0f);
  bnstats(bufA, N, 64, 64);
  bnapply(bufA, b3g1, b3b1, nullptr, bufX, N, 64, 64);          // h1
  conv(bufX, P(iB3c2), bufA, N, 64, 8, 8, 64, 3, 1, 1, 1.0f);
  conv(bufB, P(iB3sc), bufC, N, 32, 16, 16, 64, 1, 2, 0, 1.0f); // shortcut
  bnstats(bufC, N, 64, 64);
  bnapply(bufC, b3scg, b3scb, nullptr, bufC, N, 64, 64);
  bnstats(bufA, N, 64, 64);
  bnapply(bufA, b3g2, b3b2, bufC, bufB, N, 64, 64);             // bufB = out3

  // ---- pool + FC (WMMA) ----
  avgpool_kernel<<<dim3((2048 + 255) / 256), dim3(256), 0, stream>>>(
      bufB, pooled, 2048, 64);
  fc_wmma_kernel<<<dim3(1), dim3(64), 0, stream>>>(
      pooled, P(iFw), P(iFb), (float*)d_out);
}
